// MambaPolicy_13907104104646
// MI455X (gfx1250) — matrix operand order repaired and checked by reading, compiled, not run on hardware
//
#include <hip/hip_runtime.h>
#include <cstdint>
#include <cstddef>

// ---------------- types for WMMA operands ----------------
typedef __bf16 bf16_t;
typedef bf16_t v16bf __attribute__((ext_vector_type(16)));
typedef bf16_t v8bf  __attribute__((ext_vector_type(8)));
typedef float  v8f   __attribute__((ext_vector_type(8)));
typedef float  v2f   __attribute__((ext_vector_type(2)));

union ABbf { v16bf v; v8bf h[2]; };

// ---------------- problem constants ----------------
#define BATCH 4
#define SEQ   2048
#define BL    8192            // BATCH*SEQ
#define DM    1024            // d_model
#define EPROJ 4368            // d_in_proj = 2*2048 + 256 + 16
#define DI    2048            // d_inner
#define CONVD 2304            // d_inner + 2*128
#define NH    16
#define HD    128
#define DSTATE 128
#define CS    256
#define NC    8

__device__ __forceinline__ float fsilu(float x) { return x / (1.f + __expf(-x)); }

// scalar (SGPR) wave id: guarantees uniform control flow around WMMA
__device__ __forceinline__ int wave_id() {
  return __builtin_amdgcn_readfirstlane((int)(blockIdx.x * 8 + (threadIdx.x >> 5)));
}

// ---------------- f32 -> bf16 conversion ----------------
__global__ void k_f32_to_bf16(const float* __restrict__ s, bf16_t* __restrict__ d, int n) {
  int i = blockIdx.x * blockDim.x + threadIdx.x;
  if (i < n) d[i] = (bf16_t)s[i];
}

// ---------------- in_proj GEMM: [8192,1024] x [4368,1024]^T ----------------
// wave = 16(M) x 64(N) tile, bf16 WMMA, K step 32
__global__ void k_gemm_in(const bf16_t* __restrict__ A, const bf16_t* __restrict__ W,
                          float* __restrict__ z, float* __restrict__ xbc,
                          float* __restrict__ dtr) {
  const int K = DM, NE = EPROJ, NT = 69;            // 69 n64-tiles (last = 16-wide dt tail)
  int wid  = wave_id();                             // 35328 waves exactly (scalar)
  int lane = threadIdx.x & 31;
  int lm = lane & 15, hi = lane >> 4;
  int mt = wid / NT, nt = wid % NT;
  int m0 = mt * 16, n0 = nt * 64;
  const bf16_t* arow = A + (size_t)(m0 + lm) * K;

  if (n0 + 64 <= NE) {                              // uniform scalar branch: full tile
    v8f acc[4]; acc[0] = {}; acc[1] = {}; acc[2] = {}; acc[3] = {};
    for (int kb = 0; kb < K; kb += 32) {
      ABbf a;
      a.h[0] = *(const v8bf*)(arow + kb + 16 * hi);
      a.h[1] = *(const v8bf*)(arow + kb + 16 * hi + 8);
      __builtin_prefetch((const void*)(arow + kb + 256), 0, 1);
      int kw = kb + 16 * hi;
#pragma unroll
      for (int s = 0; s < 4; ++s) {
        ABbf b;
        const bf16_t* wr = W + (size_t)(n0 + 16 * s + lm) * K + kw;
        b.h[0] = *(const v8bf*)(wr);
        b.h[1] = *(const v8bf*)(wr + 8);
        acc[s] = __builtin_amdgcn_wmma_f32_16x16x32_bf16(
            false, a.v, false, b.v, (short)0, acc[s], false, false);
      }
    }
    // epilogue: whole 64-tile lies in one region (boundaries 2048/4352 are mult. of 64)
    int row0 = m0 + 8 * hi;
    if (n0 < DI) {
#pragma unroll
      for (int s = 0; s < 4; ++s) {
        float* p = z + (size_t)row0 * DI + n0 + 16 * s + lm;
#pragma unroll
        for (int v = 0; v < 8; ++v) p[(size_t)v * DI] = acc[s][v];
      }
    } else {
#pragma unroll
      for (int s = 0; s < 4; ++s) {
        float* p = xbc + (size_t)row0 * CONVD + (n0 - DI) + 16 * s + lm;
#pragma unroll
        for (int v = 0; v < 8; ++v) p[(size_t)v * CONVD] = acc[s][v];
      }
    }
  } else {                                          // tail tile: n0 = 4352 -> dt_raw (16 cols)
    v8f acc = {};
    for (int kb = 0; kb < K; kb += 32) {
      ABbf a, b;
      a.h[0] = *(const v8bf*)(arow + kb + 16 * hi);
      a.h[1] = *(const v8bf*)(arow + kb + 16 * hi + 8);
      int kw = kb + 16 * hi;
      const bf16_t* wr = W + (size_t)(n0 + lm) * K + kw;
      b.h[0] = *(const v8bf*)(wr);
      b.h[1] = *(const v8bf*)(wr + 8);
      acc = __builtin_amdgcn_wmma_f32_16x16x32_bf16(
          false, a.v, false, b.v, (short)0, acc, false, false);
    }
    float* p = dtr + (size_t)(m0 + 8 * hi) * NH + lm;
#pragma unroll
    for (int v = 0; v < 8; ++v) p[(size_t)v * NH] = acc[v];
  }
}

// ---------------- depthwise conv1d (K=4, causal) + SiLU ----------------
__global__ void k_conv_silu(const float* __restrict__ xbc, const float* __restrict__ cw,
                            const float* __restrict__ cb, float* __restrict__ xc) {
  int i = blockIdx.x * 256 + threadIdx.x;   // exact: 8192*2304
  int c = i % CONVD;
  int r = i / CONVD;
  int t = r & (SEQ - 1);
  float s = cb[c];
  const float* w = cw + c * 4;
#pragma unroll
  for (int k = 0; k < 4; ++k) {
    int tt = t + k - 3;
    if (tt >= 0) s += xbc[(size_t)(r + k - 3) * CONVD + c] * w[k];
  }
  xc[i] = fsilu(s);
}

// ---------------- dt = softplus(dt_raw + bias); dA = dt * (-exp(A_log)) ----------------
__global__ void k_dt(const float* __restrict__ dtr, const float* __restrict__ dt_bias,
                     const float* __restrict__ A_log, float* __restrict__ dtg,
                     float* __restrict__ dAg) {
  int i = blockIdx.x * 256 + threadIdx.x;   // 8192*16
  int h = i & (NH - 1);
  float v = dtr[i] + dt_bias[h];
  float dt = v > 20.f ? v : log1pf(__expf(v));
  dtg[i] = dt;
  dAg[i] = dt * (-__expf(A_log[h]));
}

// ---------------- per (b,chunk,h): cumsum + decay tables ----------------
// Acs[s]=cumsum(dA), dtc[s]=dt, dw[s]=dt*exp(Acs[last]-Acs[s]), sd[s]=exp(Acs[s])
__global__ void k_cumsum(const float* __restrict__ dtg, const float* __restrict__ dAg,
                         float* __restrict__ Acs, float* __restrict__ dtc,
                         float* __restrict__ dw, float* __restrict__ sdv) {
  int idx = blockIdx.x * 256 + threadIdx.x;  // 512 = B*NC*NH
  int h = idx & 15, c = (idx >> 4) & 7, b = idx >> 7;
  int off = idx * CS;
  int rbase = b * SEQ + c * CS;
  float acc = 0.f;
  for (int s = 0; s < CS; ++s) {
    int r = rbase + s;
    acc += dAg[r * NH + h];
    Acs[off + s] = acc;
    dtc[off + s] = dtg[r * NH + h];
  }
  float alast = acc;
  for (int s = 0; s < CS; ++s) {
    float a = Acs[off + s];
    dw[off + s]  = dtc[off + s] * __expf(alast - a);
    sdv[off + s] = __expf(a);
  }
}

// ---------------- G[l,s] = C[l] . B[s] per (b,chunk) : f32 WMMA ----------------
__global__ void k_G(const float* __restrict__ xc, float* __restrict__ G) {
  int wid  = wave_id();                            // 8192 waves (scalar)
  int lane = threadIdx.x & 31, lm = lane & 15, hi = lane >> 4;
  int bc = wid >> 8, tile = wid & 255;
  int lt = tile >> 4, st = tile & 15;
  int b = bc >> 3, c = bc & 7;
  int rowbase = b * SEQ + c * CS;
  const float* crow = xc + (size_t)(rowbase + lt * 16 + lm) * CONVD + DI + DSTATE; // C
  const float* brow = xc + (size_t)(rowbase + st * 16 + lm) * CONVD + DI;          // B
  v8f acc = {};
  for (int kb = 0; kb < DSTATE; kb += 4) {
    int k0 = kb + 2 * hi;
    v2f av = *(const v2f*)(crow + k0);
    v2f bv = *(const v2f*)(brow + k0);
    acc = __builtin_amdgcn_wmma_f32_16x16x4_f32(false, av, false, bv, (short)0, acc,
                                                false, false);
  }
  float* gp = G + (size_t)bc * (CS * CS) + (size_t)(lt * 16 + 8 * hi) * CS + st * 16 + lm;
#pragma unroll
  for (int v = 0; v < 8; ++v) gp[(size_t)v * CS] = acc[v];
}

// ---------------- Y_diag = (G * L) x xd per (b,chunk,h) : f32 WMMA ----------------
__global__ void k_Ydiag(const float* __restrict__ xc, const float* __restrict__ G,
                        const float* __restrict__ Acs, const float* __restrict__ dtc,
                        float* __restrict__ Y) {
  int wid  = wave_id();                            // 65536 waves (scalar)
  int lane = threadIdx.x & 31, lm = lane & 15, hi = lane >> 4;
  int inst = wid >> 7, tile = wid & 127;
  int lt = tile >> 3, pt = tile & 7;
  int b = inst >> 7, c = (inst >> 4) & 7, h = inst & 15;
  int off = inst * CS;
  const float* AcsP = Acs + off;
  const float* dtcP = dtc + off;
  int l = lt * 16 + lm;
  float acsl = AcsP[l];
  const float* Grow = G + (size_t)(b * 8 + c) * (CS * CS) + (size_t)l * CS;
  int rowbase = b * SEQ + c * CS;
  const float* xcol = xc + (size_t)rowbase * CONVD + h * HD + pt * 16 + lm;
  v8f acc = {};
  int kmax = lt * 16 + 16;                  // lower-triangular: skip dead K (scalar bound)
  for (int kb = 0; kb < kmax; kb += 4) {
    int s0 = kb + 2 * hi;
    v2f g  = *(const v2f*)(Grow + s0);
    v2f a2 = *(const v2f*)(AcsP + s0);
    v2f d2 = *(const v2f*)(dtcP + s0);
    v2f av, bv;
    av.x = (s0     <= l) ? g.x * __expf(acsl - a2.x) : 0.f;
    av.y = (s0 + 1 <= l) ? g.y * __expf(acsl - a2.y) : 0.f;
    bv.x = xcol[(size_t)s0 * CONVD] * d2.x;
    bv.y = xcol[(size_t)(s0 + 1) * CONVD] * d2.y;
    acc = __builtin_amdgcn_wmma_f32_16x16x4_f32(false, av, false, bv, (short)0, acc,
                                                false, false);
  }
  float* yp = Y + (size_t)(rowbase + lt * 16 + 8 * hi) * DI + h * HD + pt * 16 + lm;
#pragma unroll
  for (int v = 0; v < 8; ++v) yp[(size_t)v * DI] = acc[v];
}

// ---------------- states[p,n] = sum_s xd[s,p]*dw[s]*B[s,n] : f32 WMMA ----------------
__global__ void k_states(const float* __restrict__ xc, const float* __restrict__ dw,
                         float* __restrict__ S) {
  int wid  = wave_id();                            // 32768 waves (scalar)
  int lane = threadIdx.x & 31, lm = lane & 15, hi = lane >> 4;
  int inst = wid >> 6, tile = wid & 63;
  int pt = tile >> 3, nt = tile & 7;
  int b = inst >> 7, c = (inst >> 4) & 7, h = inst & 15;
  const float* dwP = dw + inst * CS;
  int rowbase = b * SEQ + c * CS;
  const float* xcolA = xc + (size_t)rowbase * CONVD + h * HD + pt * 16 + lm; // p row
  const float* bcol  = xc + (size_t)rowbase * CONVD + DI + nt * 16 + lm;    // n col
  v8f acc = {};
  for (int kb = 0; kb < CS; kb += 4) {
    int s0 = kb + 2 * hi;
    v2f w2 = *(const v2f*)(dwP + s0);
    v2f av, bv;
    av.x = xcolA[(size_t)s0 * CONVD] * w2.x;
    av.y = xcolA[(size_t)(s0 + 1) * CONVD] * w2.y;
    bv.x = bcol[(size_t)s0 * CONVD];
    bv.y = bcol[(size_t)(s0 + 1) * CONVD];
    acc = __builtin_amdgcn_wmma_f32_16x16x4_f32(false, av, false, bv, (short)0, acc,
                                                false, false);
  }
  float* sp = S + (size_t)inst * (HD * DSTATE) + (size_t)(pt * 16 + 8 * hi) * DSTATE +
              nt * 16 + lm;
#pragma unroll
  for (int v = 0; v < 8; ++v) sp[(size_t)v * DSTATE] = acc[v];
}

// ---------------- inter-chunk scan: prev[c+1] = prev[c]*cd[c] + states[c] ----------------
__global__ void k_scan(const float* __restrict__ S, const float* __restrict__ sdv,
                       float* __restrict__ P) {
  int bh = blockIdx.x;          // 64 = B*NH
  int b = bh >> 4, h = bh & 15;
  for (int e = threadIdx.x; e < HD * DSTATE; e += 256) {
    float carry = 0.f;
    for (int c = 0; c < NC; ++c) {
      int inst = (b * 8 + c) * 16 + h;
      size_t idx = (size_t)inst * (HD * DSTATE) + e;
      P[idx] = carry;
      float cd = sdv[inst * CS + (CS - 1)];
      carry = carry * cd + S[idx];
    }
  }
}

// ---------------- Y += (C x prev^T)*sd[l] + x*D : f32 WMMA, fused epilogue ----------------
__global__ void k_Yoff(const float* __restrict__ xc, const float* __restrict__ P,
                       const float* __restrict__ sdv, const float* __restrict__ Dv,
                       float* __restrict__ Y) {
  int wid  = wave_id();                            // 65536 waves (scalar)
  int lane = threadIdx.x & 31, lm = lane & 15, hi = lane >> 4;
  int inst = wid >> 7, tile = wid & 127;
  int lt = tile >> 3, pt = tile & 7;
  int b = inst >> 7, c = (inst >> 4) & 7, h = inst & 15;
  int rowbase = b * SEQ + c * CS;
  const float* crow = xc + (size_t)(rowbase + lt * 16 + lm) * CONVD + DI + DSTATE;
  const float* prow = P + (size_t)inst * (HD * DSTATE) + (size_t)(pt * 16 + lm) * DSTATE;
  v8f acc = {};
  for (int kb = 0; kb < DSTATE; kb += 4) {
    int k0 = kb + 2 * hi;
    v2f av = *(const v2f*)(crow + k0);
    v2f bv = *(const v2f*)(prow + k0);
    acc = __builtin_amdgcn_wmma_f32_16x16x4_f32(false, av, false, bv, (short)0, acc,
                                                false, false);
  }
  const float* sdP = sdv + inst * CS;
  float Dh = Dv[h];
  float* yp = Y + (size_t)(rowbase + lt * 16 + 8 * hi) * DI + h * HD + pt * 16 + lm;
  const float* xp = xc + (size_t)(rowbase + lt * 16 + 8 * hi) * CONVD + h * HD + pt * 16 + lm;
#pragma unroll
  for (int v = 0; v < 8; ++v) {
    int l = lt * 16 + 8 * hi + v;
    yp[(size_t)v * DI] += acc[v] * sdP[l] + xp[(size_t)v * CONVD] * Dh;
  }
}

// ---------------- gate (y * silu(z)) + RMSNorm -> bf16 ----------------
__global__ void k_gate_norm(const float* __restrict__ Y, const float* __restrict__ z,
                            const float* __restrict__ nw, bf16_t* __restrict__ yg) {
  __shared__ float red[256];
  int r = blockIdx.x, tid = threadIdx.x;
  float yv[8];
  float ss = 0.f;
#pragma unroll
  for (int i = 0; i < 8; ++i) {
    int e = i * 256 + tid;
    float g = Y[(size_t)r * DI + e] * fsilu(z[(size_t)r * DI + e]);
    yv[i] = g;
    ss += g * g;
  }
  red[tid] = ss;
  __syncthreads();
  for (int st = 128; st > 0; st >>= 1) {
    if (tid < st) red[tid] += red[tid + st];
    __syncthreads();
  }
  float scale = rsqrtf(red[0] * (1.f / DI) + 1e-5f);
#pragma unroll
  for (int i = 0; i < 8; ++i) {
    int e = i * 256 + tid;
    yg[(size_t)r * DI + e] = (bf16_t)(yv[i] * scale * nw[e]);
  }
}

// ---------------- out_proj GEMM: [8192,2048] x [1024,2048]^T -> d_out ----------------
__global__ void k_gemm_out(const bf16_t* __restrict__ A, const bf16_t* __restrict__ W,
                           float* __restrict__ out) {
  const int K = DI, NO = DM;
  int wid  = wave_id();                            // 8192 waves exactly (scalar)
  int lane = threadIdx.x & 31, lm = lane & 15, hi = lane >> 4;
  int mt = wid >> 4, nt = wid & 15;
  int m0 = mt * 16, n0 = nt * 64;
  v8f acc[4]; acc[0] = {}; acc[1] = {}; acc[2] = {}; acc[3] = {};
  const bf16_t* arow = A + (size_t)(m0 + lm) * K;
  for (int kb = 0; kb < K; kb += 32) {
    ABbf a;
    a.h[0] = *(const v8bf*)(arow + kb + 16 * hi);
    a.h[1] = *(const v8bf*)(arow + kb + 16 * hi + 8);
    __builtin_prefetch((const void*)(arow + kb + 256), 0, 1);
    int kw = kb + 16 * hi;
#pragma unroll
    for (int s = 0; s < 4; ++s) {
      ABbf b;
      const bf16_t* wr = W + (size_t)(n0 + 16 * s + lm) * K + kw;
      b.h[0] = *(const v8bf*)(wr);
      b.h[1] = *(const v8bf*)(wr + 8);
      acc[s] = __builtin_amdgcn_wmma_f32_16x16x32_bf16(
          false, a.v, false, b.v, (short)0, acc[s], false, false);
    }
  }
#pragma unroll
  for (int s = 0; s < 4; ++s) {
    float* p = out + (size_t)(m0 + 8 * hi) * NO + n0 + 16 * s + lm;
#pragma unroll
    for (int v = 0; v < 8; ++v) p[(size_t)v * NO] = acc[s][v];
  }
}

// ---------------- host launcher ----------------
extern "C" void kernel_launch(void* const* d_in, const int* in_sizes, int n_in,
                              void* d_out, int out_size, void* d_ws, size_t ws_size,
                              hipStream_t stream) {
  const float* u       = (const float*)d_in[0];
  const float* W_in    = (const float*)d_in[1];
  const float* conv_w  = (const float*)d_in[2];
  const float* conv_b  = (const float*)d_in[3];
  const float* dt_bias = (const float*)d_in[4];
  const float* A_log   = (const float*)d_in[5];
  const float* Dv      = (const float*)d_in[6];
  const float* norm_w  = (const float*)d_in[7];
  const float* W_out   = (const float*)d_in[8];
  float* out = (float*)d_out;

  char* base = (char*)d_ws;
  size_t off = 0;
  auto take = [&](size_t bytes) -> char* {
    char* p = base + off;
    off += (bytes + 255) & ~(size_t)255;
    return p;
  };
  bf16_t* ubf    = (bf16_t*)take((size_t)BL * DM * 2);
  bf16_t* winbf  = (bf16_t*)take((size_t)EPROJ * DM * 2);
  bf16_t* woutbf = (bf16_t*)take((size_t)DM * DI * 2);
  float*  zbuf   = (float*)take((size_t)BL * DI * 4);
  float*  xbcraw = (float*)take((size_t)BL * CONVD * 4);  // reused as Y after conv
  float*  xcbuf  = (float*)take((size_t)BL * CONVD * 4);
  float*  dtraw  = (float*)take((size_t)BL * NH * 4);
  float*  dtg    = (float*)take((size_t)BL * NH * 4);
  float*  dAg    = (float*)take((size_t)BL * NH * 4);
  float*  Acs    = (float*)take((size_t)512 * CS * 4);
  float*  dtc    = (float*)take((size_t)512 * CS * 4);
  float*  dwb    = (float*)take((size_t)512 * CS * 4);
  float*  sdb    = (float*)take((size_t)512 * CS * 4);
  float*  Gbuf   = (float*)take((size_t)32 * CS * CS * 4);
  float*  Sbuf   = (float*)take((size_t)512 * HD * DSTATE * 4);
  float*  Pbuf   = (float*)take((size_t)512 * HD * DSTATE * 4);
  bf16_t* ygbf   = (bf16_t*)take((size_t)BL * DI * 2);
  float*  Ybuf   = xbcraw;  // safe reuse: raw xBC dead after conv

  // 1) fp32 -> bf16 operand conversion
  k_f32_to_bf16<<<(BL * DM + 255) / 256, 256, 0, stream>>>(u, ubf, BL * DM);
  k_f32_to_bf16<<<(EPROJ * DM + 255) / 256, 256, 0, stream>>>(W_in, winbf, EPROJ * DM);
  k_f32_to_bf16<<<(DM * DI + 255) / 256, 256, 0, stream>>>(W_out, woutbf, DM * DI);
  // 2) in-projection (bf16 WMMA), split outputs
  k_gemm_in<<<4416, 256, 0, stream>>>(ubf, winbf, zbuf, xbcraw, dtraw);
  // 3) causal depthwise conv + SiLU
  k_conv_silu<<<(BL * CONVD) / 256, 256, 0, stream>>>(xbcraw, conv_w, conv_b, xcbuf);
  // 4) dt softplus / dA
  k_dt<<<(BL * NH) / 256, 256, 0, stream>>>(dtraw, dt_bias, A_log, dtg, dAg);
  // 5) per-chunk cumsum + decay tables
  k_cumsum<<<2, 256, 0, stream>>>(dtg, dAg, Acs, dtc, dwb, sdb);
  // 6) G = C.B^T per (b,chunk)    (f32 WMMA)
  k_G<<<1024, 256, 0, stream>>>(xcbuf, Gbuf);
  // 7) Y_diag (triangular, f32 WMMA) -> Ybuf
  k_Ydiag<<<8192, 256, 0, stream>>>(xcbuf, Gbuf, Acs, dtc, Ybuf);
  // 8) chunk states (f32 WMMA)
  k_states<<<4096, 256, 0, stream>>>(xcbuf, dwb, Sbuf);
  // 9) inter-chunk scan
  k_scan<<<64, 256, 0, stream>>>(Sbuf, sdb, Pbuf);
  // 10) Y_off + skip (f32 WMMA, fused epilogue into Ybuf)
  k_Yoff<<<8192, 256, 0, stream>>>(xcbuf, Pbuf, sdb, Dv, Ybuf);
  // 11) gating + RMSNorm -> bf16
  k_gate_norm<<<BL, 256, 0, stream>>>(Ybuf, zbuf, norm_w, ygbf);
  // 12) out-projection (bf16 WMMA) -> d_out
  k_gemm_out<<<1024, 256, 0, stream>>>(ygbf, woutbf, out);

  (void)in_sizes; (void)n_in; (void)out_size; (void)ws_size;
}
